// MultiHeadAttention_24438363914739
// MI455X (gfx1250) — compile-verified, matrix operand order checked by reading, not run on hardware
//
#include <hip/hip_runtime.h>
#include <hip/hip_bf16.h>

typedef __attribute__((ext_vector_type(16))) __bf16 v16bf;
typedef __attribute__((ext_vector_type(8)))  float  v8f;

#define D_MODEL 1024
#define HQ      64
#define NHEAD   16
#define BATCH   2
#define SEQ     2048
#define M_TOT   (BATCH * SEQ)      // 4096
#define N_QKV   (3 * NHEAD * HQ)   // 3072
#define SM_SCALE 0.022097086912079608f  // 1/sqrt(2048) -- reference scales by seq len!

// float -> bf16, round-to-nearest-even (pure integer ops, always compiles)
__device__ __forceinline__ __bf16 f2bf(float f) {
  unsigned int u = __float_as_uint(f);
  unsigned int r = (u + 0x7FFFu + ((u >> 16) & 1u)) >> 16;
  unsigned short s = (unsigned short)r;
  __bf16 b;
  __builtin_memcpy(&b, &s, 2);
  return b;
}

__device__ __forceinline__ v8f wmma_bf16(v16bf a, v16bf b, v8f c) {
  // (neg_a, A, neg_b, B, c_mod, C, reuse_a, reuse_b)
  return __builtin_amdgcn_wmma_f32_16x16x32_bf16(false, a, false, b, (short)0, c,
                                                 false, false);
}

__device__ __forceinline__ v16bf ldfrag(const __bf16* p) {
  return *(const v16bf*)p;
}

// Wave-local LDS fence: P staging buffer is per-wave, LDS ops are in-order
// per wave, so a dscnt drain (CDNA5 split counter) replaces s_barrier.
__device__ __forceinline__ void wave_lds_fence() {
  __builtin_amdgcn_wave_barrier();
  asm volatile("s_wait_dscnt 0x0" ::: "memory");
  __builtin_amdgcn_wave_barrier();
}

// ---------------------------------------------------------------- converts
__global__ void cvt_bf16_kernel(const float* __restrict__ in,
                                __bf16* __restrict__ out, int n) {
  for (int i = blockIdx.x * blockDim.x + threadIdx.x; i < n;
       i += gridDim.x * blockDim.x)
    out[i] = f2bf(in[i]);
}

// in[batch][R][C] (f32) -> out[batch][C][R] (bf16)
__global__ void cvt_transpose_kernel(const float* __restrict__ in,
                                     __bf16* __restrict__ out,
                                     int batch, int R, int C) {
  int total = batch * R * C;
  for (int i = blockIdx.x * blockDim.x + threadIdx.x; i < total;
       i += gridDim.x * blockDim.x) {
    int bt = i / (R * C);
    int rem = i - bt * (R * C);
    int r = rem / C;
    int c = rem - r * C;
    out[(size_t)bt * R * C + (size_t)c * R + r] = f2bf(in[i]);
  }
}

// ------------------------------------------------- fused QKV projection GEMM
// X[4096][1024] bf16  @  WqkvT[3072][1024] bf16 (N-major, K contiguous)
// wave tile 32x64 (2 M-tiles x 4 N-tiles), register double-buffered k-loop.
// n = proj*1024 + h*64 + q ; Q,K -> [b,h,seq,64] ; V -> [b,h,64,seq]
__global__ void gemm_qkv_kernel(const __bf16* __restrict__ X,
                                const __bf16* __restrict__ Wt,
                                const float* __restrict__ bq,
                                const float* __restrict__ bk,
                                const float* __restrict__ bv,
                                __bf16* __restrict__ Qb,
                                __bf16* __restrict__ Kb,
                                __bf16* __restrict__ Vt) {
  const int lane = threadIdx.x & 31;
  const int wave = threadIdx.x >> 5;
  const int hlo = lane & 15, hhi = lane >> 4;
  const int mbase = blockIdx.x * 128 + wave * 32;
  const int nbase = blockIdx.y * 64;

  const __bf16* Ap0 = X + (size_t)(mbase + hlo) * D_MODEL + hhi * 16;
  const __bf16* Ap1 = Ap0 + (size_t)16 * D_MODEL;
  const __bf16* Bp[4];
#pragma unroll
  for (int t = 0; t < 4; ++t)
    Bp[t] = Wt + (size_t)(nbase + t * 16 + hlo) * D_MODEL + hhi * 16;

  v8f acc[2][4] = {};
  // ---- prologue: fragments for k=0
  v16bf a0c = ldfrag(Ap0), a1c = ldfrag(Ap1);
  v16bf bc[4];
#pragma unroll
  for (int t = 0; t < 4; ++t) bc[t] = ldfrag(Bp[t]);

  // ---- pipelined main loop: issue k+32 loads before k's WMMAs
#pragma unroll 2
  for (int k = 0; k < D_MODEL - 32; k += 32) {
    v16bf a0n = ldfrag(Ap0 + k + 32);
    v16bf a1n = ldfrag(Ap1 + k + 32);
    v16bf bn[4];
#pragma unroll
    for (int t = 0; t < 4; ++t) bn[t] = ldfrag(Bp[t] + k + 32);
    if (k + 128 < D_MODEL) {
#pragma unroll
      for (int t = 0; t < 4; ++t) __builtin_prefetch(Bp[t] + k + 128, 0, 1);
    }
#pragma unroll
    for (int t = 0; t < 4; ++t) {
      acc[0][t] = wmma_bf16(a0c, bc[t], acc[0][t]);
      acc[1][t] = wmma_bf16(a1c, bc[t], acc[1][t]);
    }
    a0c = a0n; a1c = a1n;
#pragma unroll
    for (int t = 0; t < 4; ++t) bc[t] = bn[t];
  }
  // ---- epilogue WMMAs
#pragma unroll
  for (int t = 0; t < 4; ++t) {
    acc[0][t] = wmma_bf16(a0c, bc[t], acc[0][t]);
    acc[1][t] = wmma_bf16(a1c, bc[t], acc[1][t]);
  }

  // ---- store with Q/K/V routing + bias
#pragma unroll
  for (int i = 0; i < 2; ++i) {
#pragma unroll
    for (int t = 0; t < 4; ++t) {
#pragma unroll
      for (int r = 0; r < 8; ++r) {
        int n = nbase + t * 16 + hlo;
        int proj = n >> 10;
        int hh = (n >> 6) & 15;
        int q = n & 63;
        int m = mbase + i * 16 + r + hhi * 8;
        int b = m >> 11;
        int mk = m & 2047;
        const float* bias = (proj == 0) ? bq : (proj == 1) ? bk : bv;
        float val = acc[i][t][r] + bias[hh * HQ + q];
        if (proj == 2)
          Vt[(((size_t)b * NHEAD + hh) * HQ + q) * SEQ + mk] = f2bf(val);
        else {
          __bf16* dst = (proj == 0) ? Qb : Kb;
          dst[(((size_t)b * NHEAD + hh) * SEQ + mk) * HQ + q] = f2bf(val);
        }
      }
    }
  }
}

// --------------------------------------------------------- flash attention
// block: 128 thr (4 waves); 64 query rows / block; key chunks of 32.
// K fragments preloaded one chunk ahead; V loads issued before softmax VALU.
__global__ void attn_kernel(const __bf16* __restrict__ Qb,
                            const __bf16* __restrict__ Kb,
                            const __bf16* __restrict__ Vt,
                            __bf16* __restrict__ Attn) {
  const int lane = threadIdx.x & 31;
  const int wave = threadIdx.x >> 5;
  const int hlo = lane & 15, hhi = lane >> 4;
  const int bh = blockIdx.y;
  const int b = bh >> 4;
  const int h = bh & 15;
  const int qrow0 = blockIdx.x * 64 + wave * 16;

  const __bf16* Qp = Qb + ((size_t)bh * SEQ + qrow0) * HQ;
  const __bf16* Kp = Kb + (size_t)bh * SEQ * HQ;
  const __bf16* Vp = Vt + (size_t)bh * HQ * SEQ;

  // Q A-fragments for this wave's 16 rows (K-dim 0..31 and 32..63)
  v16bf qa0 = ldfrag(Qp + hlo * HQ + hhi * 16);
  v16bf qa1 = ldfrag(Qp + hlo * HQ + 32 + hhi * 16);

  v8f o0 = {}, o1 = {}, o2 = {}, o3 = {};
  float rmax[8], rsum[8];
#pragma unroll
  for (int r = 0; r < 8; ++r) { rmax[r] = -1e30f; rsum[r] = 0.0f; }

  __shared__ __align__(32) __bf16 pbuf[4][16 * 32];  // per-wave P staging

  // preload K fragments for chunk m0 = 0
  const __bf16* kp = Kp + (size_t)hlo * HQ + hhi * 16;
  v16bf kb00 = ldfrag(kp);
  v16bf kb01 = ldfrag(kp + 32);
  v16bf kb10 = ldfrag(kp + (size_t)16 * HQ);
  v16bf kb11 = ldfrag(kp + (size_t)16 * HQ + 32);

  for (int m0 = 0; m0 < SEQ; m0 += 32) {
    // ---- issue V loads for this chunk early (cover softmax latency)
    const __bf16* vp = Vp + (size_t)hlo * SEQ + m0 + hhi * 16;
    v16bf vb0 = ldfrag(vp);
    v16bf vb1 = ldfrag(vp + (size_t)16 * SEQ);
    v16bf vb2 = ldfrag(vp + (size_t)32 * SEQ);
    v16bf vb3 = ldfrag(vp + (size_t)48 * SEQ);

    // ---- S = Q * K^T for 32 keys (two 16x16 tiles), K frags already resident
    v8f s0 = {}, s1 = {};
    s0 = wmma_bf16(qa0, kb00, s0);
    s0 = wmma_bf16(qa1, kb01, s0);
    s1 = wmma_bf16(qa0, kb10, s1);
    s1 = wmma_bf16(qa1, kb11, s1);

    // ---- prefetch next chunk's K fragments (in flight across softmax+PV)
    v16bf nk00, nk01, nk10, nk11;
    const bool more = (m0 + 32) < SEQ;
    if (more) {
      const __bf16* kpn = Kp + (size_t)(m0 + 32 + hlo) * HQ + hhi * 16;
      nk00 = ldfrag(kpn);
      nk01 = ldfrag(kpn + 32);
      nk10 = ldfrag(kpn + (size_t)16 * HQ);
      nk11 = ldfrag(kpn + (size_t)16 * HQ + 32);
    }

    // ---- online softmax (row = r + 8*hhi lives across one 16-lane group)
    float cmax[8];
#pragma unroll
    for (int r = 0; r < 8; ++r) {
      s0[r] *= SM_SCALE;
      s1[r] *= SM_SCALE;
      cmax[r] = fmaxf(s0[r], s1[r]);
    }
#pragma unroll
    for (int mask = 1; mask < 16; mask <<= 1)
#pragma unroll
      for (int r = 0; r < 8; ++r)
        cmax[r] = fmaxf(cmax[r], __shfl_xor(cmax[r], mask, 32));

    float alpha[8], csum[8];
#pragma unroll
    for (int r = 0; r < 8; ++r) {
      float nm = fmaxf(rmax[r], cmax[r]);
      alpha[r] = __expf(rmax[r] - nm);
      rmax[r] = nm;
      float p0 = __expf(s0[r] - nm);
      float p1 = __expf(s1[r] - nm);
      csum[r] = p0 + p1;
      int row = r + hhi * 8;
      pbuf[wave][row * 32 + hlo] = f2bf(p0);
      pbuf[wave][row * 32 + 16 + hlo] = f2bf(p1);
    }
#pragma unroll
    for (int mask = 1; mask < 16; mask <<= 1)
#pragma unroll
      for (int r = 0; r < 8; ++r) csum[r] += __shfl_xor(csum[r], mask, 32);
#pragma unroll
    for (int r = 0; r < 8; ++r) {
      rsum[r] = rsum[r] * alpha[r] + csum[r];
      o0[r] *= alpha[r]; o1[r] *= alpha[r];
      o2[r] *= alpha[r]; o3[r] *= alpha[r];
    }

    // ---- wave-local fence: P stores -> A-fragment reload (same wave only)
    wave_lds_fence();
    v16bf pa = ldfrag(&pbuf[wave][hlo * 32 + hhi * 16]);

    // ---- O += P * V
    o0 = wmma_bf16(pa, vb0, o0);
    o1 = wmma_bf16(pa, vb1, o1);
    o2 = wmma_bf16(pa, vb2, o2);
    o3 = wmma_bf16(pa, vb3, o3);

    if (more) { kb00 = nk00; kb01 = nk01; kb10 = nk10; kb11 = nk11; }
  }

  // ---- normalize + store concatenated heads [b][m][h*64 + v] (bf16)
#pragma unroll
  for (int r = 0; r < 8; ++r) {
    float inv = 1.0f / rsum[r];
    int m = qrow0 + r + hhi * 8;
    __bf16* dst = Attn + ((size_t)b * SEQ + m) * (NHEAD * HQ) + h * HQ;
    dst[0 * 16 + hlo] = f2bf(o0[r] * inv);
    dst[1 * 16 + hlo] = f2bf(o1[r] * inv);
    dst[2 * 16 + hlo] = f2bf(o2[r] * inv);
    dst[3 * 16 + hlo] = f2bf(o3[r] * inv);
  }
}

// --------------------------------------------------------- output projection
// Attn[4096][1024] bf16 @ WoT[1024][1024] bf16 + bo -> f32 out
// Same 32x64 wave tile + double-buffered pipeline as the QKV GEMM.
__global__ void gemm_out_kernel(const __bf16* __restrict__ A,
                                const __bf16* __restrict__ WoT,
                                const float* __restrict__ bo,
                                float* __restrict__ Out) {
  const int lane = threadIdx.x & 31;
  const int wave = threadIdx.x >> 5;
  const int hlo = lane & 15, hhi = lane >> 4;
  const int mbase = blockIdx.x * 128 + wave * 32;
  const int nbase = blockIdx.y * 64;

  const __bf16* Ap0 = A + (size_t)(mbase + hlo) * D_MODEL + hhi * 16;
  const __bf16* Ap1 = Ap0 + (size_t)16 * D_MODEL;
  const __bf16* Bp[4];
#pragma unroll
  for (int t = 0; t < 4; ++t)
    Bp[t] = WoT + (size_t)(nbase + t * 16 + hlo) * D_MODEL + hhi * 16;

  v8f acc[2][4] = {};
  v16bf a0c = ldfrag(Ap0), a1c = ldfrag(Ap1);
  v16bf bc[4];
#pragma unroll
  for (int t = 0; t < 4; ++t) bc[t] = ldfrag(Bp[t]);

#pragma unroll 2
  for (int k = 0; k < D_MODEL - 32; k += 32) {
    v16bf a0n = ldfrag(Ap0 + k + 32);
    v16bf a1n = ldfrag(Ap1 + k + 32);
    v16bf bn[4];
#pragma unroll
    for (int t = 0; t < 4; ++t) bn[t] = ldfrag(Bp[t] + k + 32);
    if (k + 128 < D_MODEL) {
#pragma unroll
      for (int t = 0; t < 4; ++t) __builtin_prefetch(Bp[t] + k + 128, 0, 1);
    }
#pragma unroll
    for (int t = 0; t < 4; ++t) {
      acc[0][t] = wmma_bf16(a0c, bc[t], acc[0][t]);
      acc[1][t] = wmma_bf16(a1c, bc[t], acc[1][t]);
    }
    a0c = a0n; a1c = a1n;
#pragma unroll
    for (int t = 0; t < 4; ++t) bc[t] = bn[t];
  }
#pragma unroll
  for (int t = 0; t < 4; ++t) {
    acc[0][t] = wmma_bf16(a0c, bc[t], acc[0][t]);
    acc[1][t] = wmma_bf16(a1c, bc[t], acc[1][t]);
  }

#pragma unroll
  for (int i = 0; i < 2; ++i)
#pragma unroll
    for (int t = 0; t < 4; ++t)
#pragma unroll
      for (int r = 0; r < 8; ++r) {
        int n = nbase + t * 16 + hlo;
        int m = mbase + i * 16 + r + hhi * 8;
        Out[(size_t)m * D_MODEL + n] = acc[i][t][r] + bo[n];
      }
}

// ------------------------------------------------------------------ launch
extern "C" void kernel_launch(void* const* d_in, const int* in_sizes, int n_in,
                              void* d_out, int out_size, void* d_ws,
                              size_t ws_size, hipStream_t stream) {
  (void)in_sizes; (void)n_in; (void)out_size; (void)ws_size;
  const float* xi = (const float*)d_in[0];
  const float* Wq = (const float*)d_in[1];
  const float* bq = (const float*)d_in[2];
  const float* Wk = (const float*)d_in[3];
  const float* bk = (const float*)d_in[4];
  const float* Wv = (const float*)d_in[5];
  const float* bv = (const float*)d_in[6];
  const float* Wo = (const float*)d_in[7];
  const float* bo = (const float*)d_in[8];
  float* out = (float*)d_out;

  // bf16 workspace layout (~48 MB)
  __bf16* X     = (__bf16*)d_ws;                               // 4096*1024
  __bf16* WqkvT = X + (size_t)M_TOT * D_MODEL;                 // 3072*1024
  __bf16* Qb    = WqkvT + (size_t)N_QKV * D_MODEL;             // 2*16*2048*64
  __bf16* Kb    = Qb + (size_t)BATCH * NHEAD * SEQ * HQ;
  __bf16* Vt    = Kb + (size_t)BATCH * NHEAD * SEQ * HQ;       // V transposed
  __bf16* Attn  = Vt + (size_t)BATCH * NHEAD * SEQ * HQ;       // 4096*1024
  __bf16* WoT   = Attn + (size_t)M_TOT * D_MODEL;              // 1024*1024

  cvt_bf16_kernel<<<2048, 256, 0, stream>>>(xi, X, M_TOT * D_MODEL);
  cvt_transpose_kernel<<<1024, 256, 0, stream>>>(Wq, WqkvT, NHEAD, D_MODEL, HQ);
  cvt_transpose_kernel<<<1024, 256, 0, stream>>>(
      Wk, WqkvT + (size_t)NHEAD * HQ * D_MODEL, NHEAD, D_MODEL, HQ);
  cvt_transpose_kernel<<<1024, 256, 0, stream>>>(
      Wv, WqkvT + (size_t)2 * NHEAD * HQ * D_MODEL, NHEAD, D_MODEL, HQ);
  cvt_transpose_kernel<<<1024, 256, 0, stream>>>(Wo, WoT, 1, NHEAD * HQ, D_MODEL);

  dim3 gq(M_TOT / 128, N_QKV / 64);
  gemm_qkv_kernel<<<gq, 128, 0, stream>>>(X, WqkvT, bq, bk, bv, Qb, Kb, Vt);

  dim3 ga(SEQ / 64, BATCH * NHEAD);
  attn_kernel<<<ga, 128, 0, stream>>>(Qb, Kb, Vt, Attn);

  dim3 go(M_TOT / 128, D_MODEL / 64);
  gemm_out_kernel<<<go, 128, 0, stream>>>(Attn, WoT, bo, out);
}
